// MultiHeadedAttention_73461120630790
// MI455X (gfx1250) — matrix operand order repaired and checked by reading, not compiled, not run on hardware
//
#include <hip/hip_runtime.h>
#include <hip/hip_bf16.h>
#include <math.h>
#include <stddef.h>
#include <stdint.h>

typedef _Float16 half16 __attribute__((ext_vector_type(16)));
typedef _Float16 half8  __attribute__((ext_vector_type(8)));
typedef float    floatx8 __attribute__((ext_vector_type(8)));

#define WMMA_F16(a, b, c) \
  __builtin_amdgcn_wmma_f32_16x16x32_f16(false, (a), false, (b), (short)0, (c), false, false)

constexpr int Bb = 8, Ss = 1024, Dd = 768, Hh = 12, DKc = 64;
constexpr int Mtot = Bb * Ss;  // 8192

// ---------------------------------------------------------------------------
// GEMM: out = A[f32, 8192x768] @ W[f32, 768x768] + bias[768]
// OutT = _Float16 (QKV projections) or float (output projection).
// 256 thr = 8 waves, 128x128 block tile, BK=32, all dims compile-time.
// ---------------------------------------------------------------------------
template <typename OutT>
__global__ __launch_bounds__(256) void gemm_xw(
    const float* __restrict__ A, const float* __restrict__ W,
    const float* __restrict__ bias, OutT* __restrict__ out) {
  constexpr int N = 768, K = 768;
  __shared__ _Float16 As[128 * 40];  // A tile, row-major [m][k], stride 40
  __shared__ _Float16 Bt[128 * 40];  // W tile transposed [n][k], stride 40

  const int tid  = threadIdx.x;
  const int lane = tid & 31;
  const int wave = tid >> 5;
  const int ln16 = lane & 15;
  const int lh   = lane >> 4;
  const int wm   = wave >> 2;  // 0..1 : 64 rows each
  const int wn   = wave & 3;   // 0..3 : 32 cols each
  const int m0   = blockIdx.y * 128;
  const int n0   = blockIdx.x * 128;

  floatx8 acc[4][2] = {};

  for (int k0 = 0; k0 < K; k0 += 32) {
    // Prefetch next K-step tiles into cache while this step computes
    if (k0 + 32 < K) {
      __builtin_prefetch(&A[(unsigned)(m0 + (tid >> 1)) * K + k0 + 32], 0, 3);
      __builtin_prefetch(&W[(unsigned)(k0 + 32 + (tid >> 3)) * N + n0 + (tid & 7) * 16], 0, 3);
    }
    __syncthreads();
    // Stage A (convert f32 -> f16): 128x32 = 512 chunks of 8 halfs
    #pragma unroll
    for (int it = 0; it < 2; ++it) {
      int cid = tid + it * 256;
      int row = cid >> 2;
      int cb  = (cid & 3) * 8;
      const float* src = &A[(unsigned)(m0 + row) * K + k0 + cb];
      half8 h;
      #pragma unroll
      for (int j = 0; j < 8; ++j) h[j] = (_Float16)src[j];
      *(half8*)&As[row * 40 + cb] = h;
    }
    // Stage W transposed (convert f32 -> f16): W[k][n] -> Bt[n][k]
    #pragma unroll
    for (int it = 0; it < 2; ++it) {
      int cid = tid + it * 256;
      int kk  = cid >> 4;
      int nb  = (cid & 15) * 8;
      const float* src = &W[(unsigned)(k0 + kk) * N + n0 + nb];
      #pragma unroll
      for (int j = 0; j < 8; ++j) Bt[(nb + j) * 40 + kk] = (_Float16)src[j];
    }
    __syncthreads();

    // A fragments: lane ln16 = row, lh selects K octets {lh*8.., lh*8+16..}
    half16 af[4];
    #pragma unroll
    for (int mt = 0; mt < 4; ++mt) {
      int row = wm * 64 + mt * 16 + ln16;
      int kh  = lh * 8;
      half8 lo = *(const half8*)&As[row * 40 + kh];
      half8 hi = *(const half8*)&As[row * 40 + kh + 16];
      #pragma unroll
      for (int j = 0; j < 8; ++j) { af[mt][j] = lo[j]; af[mt][j + 8] = hi[j]; }
    }
    // B fragments: lane ln16 = col N; lh selects K octets {lh*8.., lh*8+16..}
    half16 bf[2];
    #pragma unroll
    for (int nt = 0; nt < 2; ++nt) {
      int n  = wn * 32 + nt * 16 + ln16;
      int kb = lh * 8;
      half8 lo = *(const half8*)&Bt[n * 40 + kb];
      half8 hi = *(const half8*)&Bt[n * 40 + kb + 16];
      #pragma unroll
      for (int j = 0; j < 8; ++j) { bf[nt][j] = lo[j]; bf[nt][j + 8] = hi[j]; }
    }
    #pragma unroll
    for (int mt = 0; mt < 4; ++mt)
      #pragma unroll
      for (int nt = 0; nt < 2; ++nt)
        acc[mt][nt] = WMMA_F16(af[mt], bf[nt], acc[mt][nt]);
  }

  // Epilogue: C layout — VGPR i holds row (lh*8 + i), col = ln16
  #pragma unroll
  for (int mt = 0; mt < 4; ++mt) {
    #pragma unroll
    for (int nt = 0; nt < 2; ++nt) {
      int col   = n0 + wn * 32 + nt * 16 + ln16;
      float bvv = bias[col];
      #pragma unroll
      for (int i = 0; i < 8; ++i) {
        int row = m0 + wm * 64 + mt * 16 + lh * 8 + i;
        out[(unsigned)row * N + col] = (OutT)(acc[mt][nt][i] + bvv);
      }
    }
  }
}

// ---------------------------------------------------------------------------
// Flash attention: one block = (b, h) x 64 Q rows; 128 thr = 4 waves x 16 rows.
// K tile staged with GLOBAL_LOAD_ASYNC_TO_LDS_B128 (ASYNCcnt); V transposed
// through VGPRs. ctxF is f32 [B,S,D] (= second output tensor).
// ---------------------------------------------------------------------------
__global__ __launch_bounds__(128) void attn_fwd(
    const _Float16* __restrict__ Q, const _Float16* __restrict__ Kp,
    const _Float16* __restrict__ Vp, float* __restrict__ ctxF) {
  __shared__ _Float16 Klds[64 * 72];      // [j][d], stride 72
  __shared__ _Float16 Vt[64 * 72];        // [d][j], stride 72 (transposed)
  __shared__ _Float16 Plds[4 * 16 * 72];  // per-wave P tile 16x64, stride 72

  const int tid  = threadIdx.x;
  const int lane = tid & 31;
  const int wave = tid >> 5;
  const int ln16 = lane & 15;
  const int lh   = lane >> 4;
  const int bh   = blockIdx.x;
  const int b    = bh / Hh;
  const int h    = bh % Hh;
  const int qr   = blockIdx.y * 64 + wave * 16;
  const float scale = 0.125f;  // 1/sqrt(64)

  const unsigned long long kbase = (unsigned long long)(uintptr_t)Kp;
  const unsigned ldsK = (unsigned)(uintptr_t)&Klds[0];  // LDS byte offset

  // Q fragments (A matrix, 16 rows x DK=64, reused across all key blocks)
  half16 qf[2];
  {
    size_t qrow = (size_t)(b * Ss + qr + ln16) * Dd + h * DKc;
    #pragma unroll
    for (int c = 0; c < 2; ++c) {
      int dbase = c * 32 + lh * 8;
      half8 lo = *(const half8*)&Q[qrow + dbase];
      half8 hi = *(const half8*)&Q[qrow + dbase + 16];
      #pragma unroll
      for (int j = 0; j < 8; ++j) { qf[c][j] = lo[j]; qf[c][j + 8] = hi[j]; }
    }
  }

  float rm[8], rl[8];
  #pragma unroll
  for (int i = 0; i < 8; ++i) { rm[i] = -1e30f; rl[i] = 0.f; }
  floatx8 oacc[4] = {};

  for (int j0 = 0; j0 < Ss; j0 += 64) {
    __syncthreads();
    // Stage K via async global->LDS (16B per lane-op), V transposed via VGPRs
    #pragma unroll
    for (int it = 0; it < 4; ++it) {
      int cid = tid + it * 128;
      int row = cid >> 3;
      int cb  = (cid & 7) * 8;
      unsigned gb = (unsigned)(((b * Ss + j0 + row) * Dd + h * DKc + cb) * 2);
      unsigned lo = ldsK + (unsigned)((row * 72 + cb) * 2);
      asm volatile("global_load_async_to_lds_b128 %0, %1, %2"
                   :: "v"(lo), "v"(gb), "s"(kbase) : "memory");
      half8 v = *(const half8*)&Vp[((size_t)(b * Ss + j0 + row) * Dd) + h * DKc + cb];
      #pragma unroll
      for (int j = 0; j < 8; ++j) Vt[(cb + j) * 72 + row] = v[j];
    }
    asm volatile("s_wait_asynccnt 0x0" ::: "memory");
    __syncthreads();

    // Scores S = Q @ K^T : 16x64 per wave; B[d][j] = K[j][d] is row j of Klds
    floatx8 sacc[4] = {};
    #pragma unroll
    for (int nt = 0; nt < 4; ++nt) {
      int j = nt * 16 + ln16;
      #pragma unroll
      for (int c = 0; c < 2; ++c) {
        int kb = c * 32 + lh * 8;
        half8 lo = *(const half8*)&Klds[j * 72 + kb];
        half8 hi = *(const half8*)&Klds[j * 72 + kb + 16];
        half16 bfrag;
        #pragma unroll
        for (int jj = 0; jj < 8; ++jj) { bfrag[jj] = lo[jj]; bfrag[jj + 8] = hi[jj]; }
        sacc[nt] = WMMA_F16(qf[c], bfrag, sacc[nt]);
      }
    }

    // Online softmax; row (lh*8+i) spans 16 lanes x 4 tiles
    _Float16* Pw = &Plds[wave * 16 * 72];
    #pragma unroll
    for (int i = 0; i < 8; ++i) {
      float m = fmaxf(fmaxf(sacc[0][i], sacc[1][i]), fmaxf(sacc[2][i], sacc[3][i]));
      m *= scale;
      #pragma unroll
      for (int d = 1; d < 16; d <<= 1) m = fmaxf(m, __shfl_xor(m, d, 32));
      float newm = fmaxf(rm[i], m);
      float corr = __expf(rm[i] - newm);
      float ps   = 0.f;
      int prow   = lh * 8 + i;
      #pragma unroll
      for (int t = 0; t < 4; ++t) {
        float p = __expf(sacc[t][i] * scale - newm);
        ps += p;
        Pw[prow * 72 + t * 16 + ln16] = (_Float16)p;
        oacc[t][i] *= corr;
      }
      #pragma unroll
      for (int d = 1; d < 16; d <<= 1) ps += __shfl_xor(ps, d, 32);
      rl[i] = rl[i] * corr + ps;
      rm[i] = newm;
    }
    __syncthreads();

    // ctx += P @ V : A = P (16 x j=64), B[j][d] read from Vt[d][j]
    #pragma unroll
    for (int c = 0; c < 2; ++c) {
      int kh = c * 32 + lh * 8;
      half8 alo = *(const half8*)&Pw[ln16 * 72 + kh];
      half8 ahi = *(const half8*)&Pw[ln16 * 72 + kh + 16];
      half16 af;
      #pragma unroll
      for (int j = 0; j < 8; ++j) { af[j] = alo[j]; af[j + 8] = ahi[j]; }
      #pragma unroll
      for (int nt = 0; nt < 4; ++nt) {
        int n  = nt * 16 + ln16;
        int kb = c * 32 + lh * 8;
        half8 blo = *(const half8*)&Vt[n * 72 + kb];
        half8 bhi = *(const half8*)&Vt[n * 72 + kb + 16];
        half16 bfrag;
        #pragma unroll
        for (int j = 0; j < 8; ++j) { bfrag[j] = blo[j]; bfrag[j + 8] = bhi[j]; }
        oacc[nt] = WMMA_F16(af, bfrag, oacc[nt]);
      }
    }
  }

  // Normalize and write f32 context
  #pragma unroll
  for (int i = 0; i < 8; ++i) {
    float inv = 1.0f / rl[i];
    int srow  = qr + lh * 8 + i;
    #pragma unroll
    for (int t = 0; t < 4; ++t) {
      int col = h * DKc + t * 16 + ln16;
      ctxF[(size_t)(b * Ss + srow) * Dd + col] = oacc[t][i] * inv;
    }
  }
}

// ---------------------------------------------------------------------------
extern "C" void kernel_launch(void* const* d_in, const int* in_sizes, int n_in,
                              void* d_out, int out_size, void* d_ws, size_t ws_size,
                              hipStream_t stream) {
  const float* query = (const float*)d_in[0];
  const float* key   = (const float*)d_in[1];
  const float* value = (const float*)d_in[2];
  const float* Wq    = (const float*)d_in[3];
  const float* bq    = (const float*)d_in[4];
  const float* Wk    = (const float*)d_in[5];
  const float* bk    = (const float*)d_in[6];
  const float* Wv    = (const float*)d_in[7];
  const float* bv    = (const float*)d_in[8];
  const float* Wd    = (const float*)d_in[9];
  const float* bd    = (const float*)d_in[10];

  float* out    = (float*)d_out;
  float* hidden = out;                      // hidden_states [B,S,D]
  float* ctx    = out + (size_t)Mtot * Dd;  // context       [B,S,D]

  _Float16* ws = (_Float16*)d_ws;
  _Float16* hQ = ws;                          // f16 projected Q [B,S,D]
  _Float16* hK = ws + (size_t)Mtot * Dd;      // f16 projected K
  _Float16* hV = ws + 2 * (size_t)Mtot * Dd;  // f16 projected V

  dim3 gg(Dd / 128, Mtot / 128);  // (6, 64)
  gemm_xw<_Float16><<<gg, 256, 0, stream>>>(query, Wq, bq, hQ);
  gemm_xw<_Float16><<<gg, 256, 0, stream>>>(key,   Wk, bk, hK);
  gemm_xw<_Float16><<<gg, 256, 0, stream>>>(value, Wv, bv, hV);

  attn_fwd<<<dim3(Bb * Hh, Ss / 64), 128, 0, stream>>>(hQ, hK, hV, ctx);

  gemm_xw<float><<<gg, 256, 0, stream>>>(ctx, Wd, bd, hidden);
}
